// StableRNN_8083128451128
// MI455X (gfx1250) — matrix operand order repaired and checked by reading, not compiled, not run on hardware
//
#include <hip/hip_runtime.h>

// ---------------------------------------------------------------------------
// StableRNN for MI455X (gfx1250, wave32, WMMA).
//   1) expm(W_log) via scaling-and-squaring (fp32, one-time, tiny)
//   2) u = x @ U_w^T + U_b          : split-bf16 WMMA GEMM (memory-bound)
//   3) serial scan (4096 steps)     : single-WGP. Whi fragments reg-resident,
//      Wlo in LDS, h state in registers, double-buffered bf16 h in LDS,
//      ONE barrier per step.
//   4) out = hidden @ c_w^T + c_b   : split-bf16 WMMA GEMM, in place in d_out
// Split-bf16: x = hi + lo, A@B ~= Ahi@Bhi + Ahi@Blo + Alo@Bhi  (~2^-16 rel err)
// ---------------------------------------------------------------------------

#define T_STEPS 4096
#define TB      65536      // T * B
#define DT_STEP 0.1f

typedef __attribute__((ext_vector_type(16))) __bf16 v16bf;
typedef __attribute__((ext_vector_type(8)))  __bf16 v8bf;
typedef __attribute__((ext_vector_type(8)))  float  v8f;

__device__ __forceinline__ v8f wmma_bf16(v16bf a, v16bf b, v8f c) {
  // (neg_a, A, neg_b, B, c_mod, C, reuse_a, reuse_b)
  return __builtin_amdgcn_wmma_f32_16x16x32_bf16(false, a, false, b, (short)0, c,
                                                 false, false);
}

__device__ __forceinline__ void split_bf16(float f, __bf16& hi, __bf16& lo) {
  hi = (__bf16)f;
  lo = (__bf16)(f - (float)hi);
}

// A-fragment (16x32 bf16): lane holds M=lane&15; K-run [16*half,+16)
// packed contiguously as vector elements 0..15.
__device__ __forceinline__ v16bf load_a_frag(const __bf16* row, int k0) {
  v8bf r0 = *(const v8bf*)(row + k0);
  v8bf r1 = *(const v8bf*)(row + k0 + 8);
  v16bf a;
#pragma unroll
  for (int i = 0; i < 8; ++i) { a[i] = r0[i]; a[8 + i] = r1[i]; }
  return a;
}

// ------------------------------ expm pieces --------------------------------

__global__ void expm_init_kernel(const float* __restrict__ Wlog,
                                 float* __restrict__ X, float* __restrict__ R) {
  int i = blockIdx.x * blockDim.x + threadIdx.x;
  if (i < 65536) {
    float xv = Wlog[i] * (1.0f / 64.0f);       // scale by 2^-6
    X[i] = xv;
    float diag = ((i % 257) == 0) ? 1.0f : 0.0f;
    R[i] = diag + xv * 0.125f;                  // innermost Horner: I + X/8
  }
}

// C = alpha*(A@B) + addI*I   (256x256 fp32, naive: one-time tiny cost)
__global__ void matmul256_kernel(const float* __restrict__ A,
                                 const float* __restrict__ B,
                                 float* __restrict__ C, float alpha, int addI) {
  int m = blockIdx.y * 16 + (threadIdx.x >> 4);
  int n = blockIdx.x * 16 + (threadIdx.x & 15);
  float acc = 0.0f;
#pragma unroll 8
  for (int k = 0; k < 256; ++k) acc = fmaf(A[m * 256 + k], B[k * 256 + n], acc);
  float r = alpha * acc;
  if (addI && (m == n)) r += 1.0f;
  C[m * 256 + n] = r;
}

__global__ void split_kernel(const float* __restrict__ src,
                             __bf16* __restrict__ hi, __bf16* __restrict__ lo,
                             int n) {
  int i = blockIdx.x * blockDim.x + threadIdx.x;
  if (i < n) {
    __bf16 h, l;
    split_bf16(src[i], h, l);
    hi[i] = h; lo[i] = l;
  }
}

// --------------------- big row-major GEMM (M x 256 x 256) ------------------
// Out[M][256] = A[M][256] @ Bt^T + bias, where BHi/BLo are B^T stored
// row-major (i.e. the original U_w / c_w, bf16 hi/lo). In-place safe
// (Out==A): all global A reads complete before the barrier.
__global__ __launch_bounds__(256) void gemm_rm256_kernel(
    const float* __restrict__ A, const __bf16* __restrict__ BHi,
    const __bf16* __restrict__ BLo, const float* __restrict__ bias,
    float* __restrict__ Out) {
  extern __shared__ char smem[];
  __bf16* aHi = (__bf16*)smem;          // [128][256]
  __bf16* aLo = aHi + 128 * 256;        // [128][256]

  const int tid = threadIdx.x;
  const long rowBase = (long)blockIdx.x * 128;
  const float* Ab = A + rowBase * 256;

  // stage + split 128 rows of A into LDS
  for (int i = tid * 4; i < 128 * 256; i += 256 * 4) {
    float4 v = *(const float4*)(Ab + i);
    __bf16 h, l;
    split_bf16(v.x, h, l); aHi[i + 0] = h; aLo[i + 0] = l;
    split_bf16(v.y, h, l); aHi[i + 1] = h; aLo[i + 1] = l;
    split_bf16(v.z, h, l); aHi[i + 2] = h; aLo[i + 2] = l;
    split_bf16(v.w, h, l); aHi[i + 3] = h; aLo[i + 3] = l;
  }
  __syncthreads();

  const int lane  = tid & 31;
  const int wave  = tid >> 5;
  const int mrow  = wave * 16;      // 8 waves x 16 rows = 128 rows
  const int lN    = lane & 15;
  const int lHalf = lane >> 4;

  const __bf16* arowHi = aHi + (mrow + lN) * 256;
  const __bf16* arowLo = aLo + (mrow + lN) * 256;

  const v8f vzero = {0.f, 0.f, 0.f, 0.f, 0.f, 0.f, 0.f, 0.f};

  for (int nc = 0; nc < 4; ++nc) {
    v8f acc[4] = {vzero, vzero, vzero, vzero};
    for (int kt = 0; kt < 8; ++kt) {
      int k0 = kt * 32 + lHalf * 16;
      v16bf afh = load_a_frag(arowHi, k0);
      v16bf afl = load_a_frag(arowLo, k0);
      int koff = kt * 32 + lHalf * 16;   // B frag: lane holds row lN of B^T
#pragma unroll
      for (int j = 0; j < 4; ++j) {
        int n0 = (nc * 4 + j) * 16;
        v16bf bfh = *(const v16bf*)(BHi + (n0 + lN) * 256 + koff);
        v16bf bfl = *(const v16bf*)(BLo + (n0 + lN) * 256 + koff);
        acc[j] = wmma_bf16(afh, bfh, acc[j]);
        acc[j] = wmma_bf16(afh, bfl, acc[j]);
        acc[j] = wmma_bf16(afl, bfh, acc[j]);
      }
    }
#pragma unroll
    for (int j = 0; j < 4; ++j) {
      int n0 = (nc * 4 + j) * 16;
      float bv = bias[n0 + lN];
#pragma unroll
      for (int r = 0; r < 8; ++r) {
        long m = rowBase + mrow + lHalf * 8 + r;   // C/D layout row
        Out[m * 256 + n0 + lN] = acc[j][r] + bv;
      }
    }
  }
}

// ------------------------------- the scan ----------------------------------
// One workgroup (8 waves), each wave owns two 16-wide N tiles.
// Whi fragments: registers (loop-invariant). Wlo: LDS (128KB).
// h fp32 state: registers (each lane re-produces the same C-layout elements
// every step). h bf16 hi/lo: double-buffered LDS -> ONE barrier per step.
__device__ __forceinline__ void rnn_step(
    int t, const float* __restrict__ u, float* __restrict__ hidden,
    const __bf16* rHi, const __bf16* rLo,         // read buffers (h_t)
    __bf16* wHi, __bf16* wLo,                     // write buffers (h_{t+1})
    const __bf16* sWLo, const v16bf* wh0, const v16bf* wh1,
    float* hprev0, float* hprev1, int lN, int lHalf, int n0a, int n0b) {
  const int mBase = lHalf * 8;

  // u_t loads issued up front (u is L2-resident: 64MB < 192MB L2)
  float uv0[8], uv1[8];
#pragma unroll
  for (int r = 0; r < 8; ++r) {
    long base = ((long)t * 16 + mBase + r) * 256;
    uv0[r] = u[base + n0a + lN];
    uv1[r] = u[base + n0b + lN];
  }

  const v8f vzero = {0.f, 0.f, 0.f, 0.f, 0.f, 0.f, 0.f, 0.f};
  v8f acc0 = vzero, acc1 = vzero;
  const __bf16* arowHi = rHi + lN * 256;
  const __bf16* arowLo = rLo + lN * 256;
#pragma unroll
  for (int kt = 0; kt < 8; ++kt) {
    int k0 = kt * 32 + lHalf * 16;
    v16bf afh = load_a_frag(arowHi, k0);
    v16bf afl = load_a_frag(arowLo, k0);
    int koff = kt * 32 + lHalf * 16;
    v16bf bl0 = *(const v16bf*)(sWLo + (n0a + lN) * 256 + koff);
    v16bf bl1 = *(const v16bf*)(sWLo + (n0b + lN) * 256 + koff);
    acc0 = wmma_bf16(afh, wh0[kt], acc0);
    acc0 = wmma_bf16(afh, bl0, acc0);
    acc0 = wmma_bf16(afl, wh0[kt], acc0);
    acc1 = wmma_bf16(afh, wh1[kt], acc1);
    acc1 = wmma_bf16(afh, bl1, acc1);
    acc1 = wmma_bf16(afl, wh1[kt], acc1);
  }

  // h_next = h + dt * tanh(u_t - h @ Wexp^T); h_old lives in registers
#pragma unroll
  for (int r = 0; r < 8; ++r) {
    int m = mBase + r;
    float hn0 = hprev0[r] + DT_STEP * tanhf(uv0[r] - acc0[r]);
    float hn1 = hprev1[r] + DT_STEP * tanhf(uv1[r] - acc1[r]);
    hprev0[r] = hn0;
    hprev1[r] = hn1;
    long gb = ((long)t * 16 + m) * 256;
    hidden[gb + n0a + lN] = hn0;
    hidden[gb + n0b + lN] = hn1;
    int i0 = m * 256 + n0a + lN;
    int i1 = m * 256 + n0b + lN;
    __bf16 hh, hl;
    split_bf16(hn0, hh, hl); wHi[i0] = hh; wLo[i0] = hl;
    split_bf16(hn1, hh, hl); wHi[i1] = hh; wLo[i1] = hl;
  }
  // reads of buf(t) precede writes of buf(t+1) in program order per wave, so
  // one barrier both publishes h_{t+1} and protects buf(t) reuse next step.
  __syncthreads();
}

__global__ __launch_bounds__(256) void scan_kernel(
    const float* __restrict__ u, const __bf16* __restrict__ gWHi,
    const __bf16* __restrict__ gWLo, float* __restrict__ hidden,
    float* __restrict__ hfin) {
  extern __shared__ char smem[];
  __bf16* sWLo  = (__bf16*)smem;        // 65536 elems (128KB)
  __bf16* sHHi0 = sWLo + 65536;         // [16][256] buf0 hi
  __bf16* sHLo0 = sHHi0 + 4096;
  __bf16* sHHi1 = sHLo0 + 4096;         // [16][256] buf1 hi
  __bf16* sHLo1 = sHHi1 + 4096;

  const int tid = threadIdx.x;
  {
    const uint4* srcL = (const uint4*)gWLo;
    uint4* dstL = (uint4*)sWLo;
    for (int i = tid; i < 8192; i += 256) dstL[i] = srcL[i];
  }
  for (int i = tid; i < 4096; i += 256) {   // zero buffer 0 (h_0 = 0)
    sHHi0[i] = (__bf16)0.0f; sHLo0[i] = (__bf16)0.0f;
  }

  const int lane  = tid & 31;
  const int wave  = tid >> 5;
  const int lN    = lane & 15;
  const int lHalf = lane >> 4;
  const int n0a   = (2 * wave) * 16;
  const int n0b   = n0a + 16;

  // loop-invariant Whi B-fragments -> registers (2 tiles x 8 k-tiles)
  v16bf wh0[8], wh1[8];
#pragma unroll
  for (int kt = 0; kt < 8; ++kt) {
    int koff = kt * 32 + lHalf * 16;
    wh0[kt] = *(const v16bf*)(gWHi + (n0a + lN) * 256 + koff);
    wh1[kt] = *(const v16bf*)(gWHi + (n0b + lN) * 256 + koff);
  }

  float hprev0[8], hprev1[8];
#pragma unroll
  for (int r = 0; r < 8; ++r) { hprev0[r] = 0.0f; hprev1[r] = 0.0f; }

  __syncthreads();

  for (int t = 0; t < T_STEPS; t += 2) {
    rnn_step(t,     u, hidden, sHHi0, sHLo0, sHHi1, sHLo1, sWLo, wh0, wh1,
             hprev0, hprev1, lN, lHalf, n0a, n0b);
    rnn_step(t + 1, u, hidden, sHHi1, sHLo1, sHHi0, sHLo0, sWLo, wh0, wh1,
             hprev0, hprev1, lN, lHalf, n0a, n0b);
  }

  // h_final from register state (positions fixed per lane)
#pragma unroll
  for (int r = 0; r < 8; ++r) {
    int m = lHalf * 8 + r;
    hfin[m * 256 + n0a + lN] = hprev0[r];
    hfin[m * 256 + n0b + lN] = hprev1[r];
  }
}

// ------------------------------- launcher ----------------------------------

extern "C" void kernel_launch(void* const* d_in, const int* in_sizes, int n_in,
                              void* d_out, int out_size, void* d_ws,
                              size_t ws_size, hipStream_t stream) {
  (void)in_sizes; (void)n_in; (void)out_size; (void)ws_size;
  const float* x    = (const float*)d_in[0];
  const float* U_w  = (const float*)d_in[1];
  const float* U_b  = (const float*)d_in[2];
  const float* Wlog = (const float*)d_in[3];
  const float* c_w  = (const float*)d_in[4];
  const float* c_b  = (const float*)d_in[5];

  float* out  = (float*)d_out;                 // [T][B][256]; holds hidden first
  float* hfin = out + (long)TB * 256;          // [16][256]

  // workspace layout: u (64MB) + expm fp32 buffers + bf16 hi/lo weights
  float*  u   = (float*)d_ws;                  // TB*256 floats
  float*  mX  = u + (long)TB * 256;            // 65536 floats (scaled W_log)
  float*  mR  = mX + 65536;
  float*  mT  = mR + 65536;
  __bf16* WHi  = (__bf16*)(mT + 65536);
  __bf16* WLo  = WHi + 65536;
  __bf16* UwHi = WLo + 65536;
  __bf16* UwLo = UwHi + 65536;
  __bf16* CwHi = UwLo + 65536;
  __bf16* CwLo = CwHi + 65536;

  // --- expm(W_log): X = W_log/64; Taylor-8 Horner; square 6 times ---
  expm_init_kernel<<<256, 256, 0, stream>>>(Wlog, mX, mR);
  float* src = mR;
  float* dst = mT;
  for (int k = 7; k >= 1; --k) {               // R = I + (X/k) @ R
    matmul256_kernel<<<dim3(16, 16), 256, 0, stream>>>(mX, src, dst,
                                                       1.0f / (float)k, 1);
    float* tmp = src; src = dst; dst = tmp;
  }
  for (int s = 0; s < 6; ++s) {                // R = R @ R
    matmul256_kernel<<<dim3(16, 16), 256, 0, stream>>>(src, src, dst, 1.0f, 0);
    float* tmp = src; src = dst; dst = tmp;
  }

  // --- bf16 hi/lo splits (row-major == B^T layout for our GEMMs) ---
  split_kernel<<<256, 256, 0, stream>>>(src, WHi, WLo, 65536);
  split_kernel<<<256, 256, 0, stream>>>(U_w, UwHi, UwLo, 65536);
  split_kernel<<<256, 256, 0, stream>>>(c_w, CwHi, CwLo, 65536);

  // --- u = x @ U_w^T + U_b  (memory-bound WMMA GEMM) ---
  gemm_rm256_kernel<<<512, 256, 131072, stream>>>(x, UwHi, UwLo, U_b, u);

  // --- serial scan: hidden -> d_out, h_final -> tail (160KB dynamic LDS) ---
  scan_kernel<<<1, 256, 163840, stream>>>(u, WHi, WLo, out, hfin);

  // --- out = hidden @ c_w^T + c_b, in place over d_out ---
  gemm_rm256_kernel<<<512, 256, 131072, stream>>>(out, CwHi, CwLo, c_b, out);
}
